// AxialAttention3D_56865366999441
// MI455X (gfx1250) — compile-verified, matrix operand order checked by reading, not run on hardware
//
#include <hip/hip_runtime.h>
#include <hip/hip_bf16.h>

// ---------------------------------------------------------------------------
// AxialAttention3D for MI455X (gfx1250): wave32 + v_wmma_f32_16x16x32_f16
// One workgroup (8 waves) per (b,d,h) row: 64x128 tile resident in LDS.
// ---------------------------------------------------------------------------

typedef __attribute__((ext_vector_type(16))) _Float16 v16h;
typedef __attribute__((ext_vector_type(8)))  _Float16 v8h;
typedef __attribute__((ext_vector_type(8)))  float    v8f;
typedef __attribute__((ext_vector_type(4)))  int      v4i;

#define NH   4
#define HD   32
#define SEQ  64      // W
#define CH   128     // C
#define C3   384     // 3C

// ---- LDS layout (bytes), buffers overlapped by lifetime -------------------
#define X_OFF     0                        // f32 x[64][128]              32768
#define QKV_OFF   32768                    // f16 qkv[64][392]            50176
#define VT_OFF    (QKV_OFF + 50176)        // f16 vT[4][32][72]           18432
#define XNAT_OFF  (VT_OFF + 18432)         // f16 xn[64][136] | attn[4][64][64] 32768
#define O_OFF     (XNAT_OFF + 32768)       // f16 o[64][136]              17408
#define LDS_BYTES (O_OFF + 17408)          // 151552 (< 320KB/WGP)

#define QKV_PITCH 392
#define XN_PITCH  136
#define O_PITCH   136
#define VT_PITCH  72

#if defined(__gfx1250__) && __has_builtin(__builtin_amdgcn_global_load_async_to_lds_b128)
#define ASYNC_LDS 1
typedef __attribute__((address_space(1))) v4i* as1_v4i;
typedef __attribute__((address_space(3))) v4i* as3_v4i;
#else
#define ASYNC_LDS 0
#endif

// ---- 16x32 f16 fragment load (A layout; B uses N-major source = same map) --
// lanes 0-15: rows M0..15, halves K[0..8)+K[16..24); lanes 16-31: K[8..16)+K[24..32)
__device__ __forceinline__ v16h load_frag16(const _Float16* base, int row0,
                                            int pitch, int col0, int lane) {
  const int lo = lane & 15;
  const int hs = (lane >> 4) & 1;
  const _Float16* p = base + (size_t)(row0 + lo) * pitch + col0 + hs * 8;
  v8h a = *(const v8h*)(p);
  v8h b = *(const v8h*)(p + 16);
  v16h r;
#pragma unroll
  for (int i = 0; i < 8; ++i) { r[i] = a[i]; r[i + 8] = b[i]; }
  return r;
}

__device__ __forceinline__ v8f wmma16(v16h a, v16h b, v8f c) {
  return __builtin_amdgcn_wmma_f32_16x16x32_f16(false, a, false, b,
                                                (short)0, c, false, false);
}

// ---- weight prep: f32 -> f16, transposed to N-major ------------------------
__global__ void prep_weights_kernel(const float* __restrict__ wqkv,
                                    const float* __restrict__ wproj,
                                    _Float16* __restrict__ wqkvT,
                                    _Float16* __restrict__ wprojT) {
  int i = blockIdx.x * blockDim.x + threadIdx.x;
  if (i < C3 * CH) {                       // wqkvT[n][c] = wqkv[c][n]
    int n = i / CH, c = i % CH;
    wqkvT[i] = (_Float16)wqkv[c * C3 + n];
  }
  int j = i - C3 * CH;
  if (j >= 0 && j < CH * CH) {             // wprojT[n][c] = wproj[c][n]
    int n = j / CH, c = j % CH;
    wprojT[j] = (_Float16)wproj[c * CH + n];
  }
}

__global__ __launch_bounds__(256)
void axial_attn_kernel(const float* __restrict__ x,
                       const float* __restrict__ gamma,
                       const float* __restrict__ beta,
                       const _Float16* __restrict__ wqkvT,
                       const _Float16* __restrict__ wprojT,
                       float* __restrict__ out) {
  __shared__ __align__(16) unsigned char lds[LDS_BYTES];
  float*    xs   = (float*)(lds + X_OFF);
  _Float16* qkvs = (_Float16*)(lds + QKV_OFF);
  _Float16* vts  = (_Float16*)(lds + VT_OFF);
  _Float16* xns  = (_Float16*)(lds + XNAT_OFF);   // lifetime: LN -> GEMM1
  _Float16* atts = (_Float16*)(lds + XNAT_OFF);   // lifetime: softmax -> GEMM3
  _Float16* os   = (_Float16*)(lds + O_OFF);

  const int tid  = threadIdx.x;
  const int lane = tid & 31;
  const int wave = tid >> 5;
  const int lo   = lane & 15;
  const int hi8  = (lane >> 4) * 8;
  const size_t nbase = (size_t)blockIdx.x * (SEQ * CH);

  // warm weights toward the caches (global_prefetch_b8)
  if (tid < 104) __builtin_prefetch(wqkvT + (size_t)tid * 512, 0, 3);

  // ---- stage 0: x tile (64x128 f32 = 32KB) -> LDS --------------------------
  {
    const float* gp = x + nbase;
#if ASYNC_LDS
#pragma unroll
    for (int i = 0; i < 8; ++i) {
      const int idx = (i * 256 + tid) * 4;
      __builtin_amdgcn_global_load_async_to_lds_b128(
          (as1_v4i)(gp + idx), (as3_v4i)(xs + idx), 0, 0);
    }
    __builtin_amdgcn_s_wait_asynccnt(0);
#else
#pragma unroll
    for (int i = 0; i < 8; ++i) {
      const int idx = (i * 256 + tid) * 4;
      *(float4*)(xs + idx) = *(const float4*)(gp + idx);
    }
#endif
  }
  __syncthreads();

  // ---- stage 1: LayerNorm (f32) -> xn f16 ----------------------------------
  {
    const float4 g4 = *(const float4*)(gamma + lane * 4);
    const float4 b4 = *(const float4*)(beta + lane * 4);
#pragma unroll
    for (int i = 0; i < 8; ++i) {
      const int r = wave * 8 + i;
      float4 v = *(const float4*)(xs + r * CH + lane * 4);
      float sum = v.x + v.y + v.z + v.w;
      float sq  = v.x * v.x + v.y * v.y + v.z * v.z + v.w * v.w;
#pragma unroll
      for (int off = 1; off < 32; off <<= 1) {
        sum += __shfl_xor(sum, off);
        sq  += __shfl_xor(sq, off);
      }
      const float mean = sum * (1.0f / CH);
      const float var  = sq * (1.0f / CH) - mean * mean;
      const float rstd = rsqrtf(var + 1e-6f);
      _Float16* xr = xns + r * XN_PITCH + lane * 4;
      xr[0] = (_Float16)((v.x - mean) * rstd * g4.x + b4.x);
      xr[1] = (_Float16)((v.y - mean) * rstd * g4.y + b4.y);
      xr[2] = (_Float16)((v.z - mean) * rstd * g4.z + b4.z);
      xr[3] = (_Float16)((v.w - mean) * rstd * g4.w + b4.w);
    }
  }
  __syncthreads();

  // ---- stage 2: QKV GEMM (64x128 @ 128x384), 96 tiles, K=4 WMMA steps ------
  // Tile indices forced scalar via readfirstlane -> SALU math, uniform branch.
  // v written transposed per head so attn@v B-fragments are contiguous.
  for (int t0 = wave; t0 < 96; t0 += 8) {
    const int t  = __builtin_amdgcn_readfirstlane(t0);
    const int mt = t / 24;
    const int nt = t % 24;
    v8f acc = {};
#pragma unroll
    for (int kk = 0; kk < 4; ++kk) {
      v16h a = load_frag16(xns, mt * 16, XN_PITCH, kk * 32, lane);
      v16h b = load_frag16(wqkvT, nt * 16, CH, kk * 32, lane);
      acc = wmma16(a, b, acc);
    }
    const int h    = nt / 6;             // head           (scalar)
    const int ntw  = nt - h * 6;         // tile in head   (scalar)
    const int part = ntw >> 1;           // 0=q 1=k 2=v    (scalar)
    if (part == 2) {
      // d = in-head channel = (ntw-4)*16 + lo ; 8 rows are contiguous in vT
      const int d = (ntw - 4) * 16 + lo;
      v8h pk;
#pragma unroll
      for (int r = 0; r < 8; ++r) pk[r] = (_Float16)acc[r];
      *(v8h*)(vts + (h * HD + d) * VT_PITCH + mt * 16 + hi8) = pk;
    } else {
      const int col = nt * 16 + lo;
#pragma unroll
      for (int r = 0; r < 8; ++r)
        qkvs[(mt * 16 + r + hi8) * QKV_PITCH + col] = (_Float16)acc[r];
    }
  }
  __syncthreads();

  // ---- stage 3: scores = q k^T * scale, softmax in registers ---------------
  // 16 row-blocks (head,stile); each wave owns whole rows -> register softmax.
  for (int rb0 = wave; rb0 < 16; rb0 += 8) {
    const int rb = __builtin_amdgcn_readfirstlane(rb0);
    const int h  = rb >> 2;
    const int st = rb & 3;
    v16h aq = load_frag16(qkvs, st * 16, QKV_PITCH, h * 96, lane);        // q
    v8f sc[4];
#pragma unroll
    for (int tt = 0; tt < 4; ++tt) {
      v16h bk = load_frag16(qkvs, tt * 16, QKV_PITCH, h * 96 + HD, lane); // k
      v8f z = {};
      sc[tt] = wmma16(aq, bk, z);
    }
    const float scale = 0.17677669529663687f;   // hd^-0.5
    _Float16* ah = atts + h * (SEQ * SEQ);
#pragma unroll
    for (int r = 0; r < 8; ++r) {
      float sv[4];
      float m = -3.0e38f;
#pragma unroll
      for (int tt = 0; tt < 4; ++tt) { sv[tt] = sc[tt][r] * scale; m = fmaxf(m, sv[tt]); }
#pragma unroll
      for (int off = 1; off < 16; off <<= 1) m = fmaxf(m, __shfl_xor(m, off));
      float e[4]; float s = 0.0f;
#pragma unroll
      for (int tt = 0; tt < 4; ++tt) { e[tt] = __expf(sv[tt] - m); s += e[tt]; }
#pragma unroll
      for (int off = 1; off < 16; off <<= 1) s += __shfl_xor(s, off);
      const float inv = 1.0f / s;
      const int row = st * 16 + r + hi8;
#pragma unroll
      for (int tt = 0; tt < 4; ++tt)
        ah[row * SEQ + tt * 16 + lo] = (_Float16)(e[tt] * inv);
    }
  }
  __syncthreads();

  // ---- stage 4: o = attn @ v (per head), 32 tiles, K=2 WMMA steps ----------
  for (int t0 = wave; t0 < 32; t0 += 8) {
    const int t  = __builtin_amdgcn_readfirstlane(t0);
    const int h  = t >> 3;
    const int st = (t >> 1) & 3;
    const int dt = t & 1;
    v8f acc = {};
#pragma unroll
    for (int kk = 0; kk < 2; ++kk) {
      v16h a = load_frag16(atts + h * SEQ * SEQ, st * 16, SEQ, kk * 32, lane);
      v16h b = load_frag16(vts + h * HD * VT_PITCH, dt * 16, VT_PITCH, kk * 32, lane);
      acc = wmma16(a, b, acc);
    }
    const int col = h * HD + dt * 16 + lo;
#pragma unroll
    for (int r = 0; r < 8; ++r)
      os[(st * 16 + r + hi8) * O_PITCH + col] = (_Float16)acc[r];
  }
  __syncthreads();

  // ---- stage 5: y = o @ w_proj + x (residual), store f32 -------------------
  for (int t0 = wave; t0 < 32; t0 += 8) {
    const int t  = __builtin_amdgcn_readfirstlane(t0);
    const int mt = t >> 3;
    const int nt = t & 7;
    v8f acc = {};
#pragma unroll
    for (int kk = 0; kk < 4; ++kk) {
      v16h a = load_frag16(os, mt * 16, O_PITCH, kk * 32, lane);
      v16h b = load_frag16(wprojT, nt * 16, CH, kk * 32, lane);
      acc = wmma16(a, b, acc);
    }
    const int col = nt * 16 + lo;
#pragma unroll
    for (int r = 0; r < 8; ++r) {
      const int row = mt * 16 + r + hi8;
      out[nbase + row * CH + col] = xs[row * CH + col] + acc[r];
    }
  }
}

extern "C" void kernel_launch(void* const* d_in, const int* in_sizes, int n_in,
                              void* d_out, int out_size, void* d_ws, size_t ws_size,
                              hipStream_t stream) {
  const float* x     = (const float*)d_in[0];
  const float* gam   = (const float*)d_in[1];
  const float* bet   = (const float*)d_in[2];
  const float* wqkv  = (const float*)d_in[3];
  const float* wproj = (const float*)d_in[4];
  float* out = (float*)d_out;

  _Float16* wqkvT  = (_Float16*)d_ws;          // 384x128 f16 =  98304 B
  _Float16* wprojT = wqkvT + C3 * CH;          // 128x128 f16 =  32768 B

  const int prep_n = C3 * CH + CH * CH;
  prep_weights_kernel<<<(prep_n + 255) / 256, 256, 0, stream>>>(wqkv, wproj,
                                                               wqkvT, wprojT);

  const int nblocks = in_sizes[0] / (SEQ * CH);   // B*D*H = 8192
  axial_attn_kernel<<<nblocks, 256, 0, stream>>>(x, gam, bet, wqkvT, wprojT, out);
}
